// HGT_policy_46437186404691
// MI455X (gfx1250) — compile-verified, matrix operand order checked by reading, not run on hardware
//
#include <hip/hip_runtime.h>
#include <hip/hip_bf16.h>
#include <math.h>

#define NA_N   40000
#define NB_N   40000
#define EDG    250000
#define CDIM   128
#define HHE    8
#define DD     16
#define LL     2

typedef __attribute__((ext_vector_type(16))) __bf16 v16bf;
typedef __attribute__((ext_vector_type(8)))  __bf16 v8bf;
typedef __attribute__((ext_vector_type(8)))  float  v8f;
typedef __attribute__((ext_vector_type(4)))  unsigned v4u;
typedef __attribute__((ext_vector_type(8)))  int      v8i;
typedef __attribute__((ext_vector_type(4)))  int      v4i;

#define GF_RELU  1
#define GF_SKIP  4

#define CAT16(lo, hi) __builtin_shufflevector(lo, hi, 0,1,2,3,4,5,6,7,8,9,10,11,12,13,14,15)

__device__ __forceinline__ float gelu_exact(float x) {
  return 0.5f * x * (1.0f + erff(x * 0.70710678118654752f));
}
// monotone float<->uint mapping for integer atomicMax-based float max
__device__ __forceinline__ unsigned ford(float f) {
  unsigned u = __float_as_uint(f);
  return (u & 0x80000000u) ? ~u : (u | 0x80000000u);
}
__device__ __forceinline__ float funord(unsigned u) {
  u = (u & 0x80000000u) ? (u & 0x7fffffffu) : ~u;
  return __uint_as_float(u);
}

__global__ void fill_u32_kernel(unsigned* p, unsigned v, long long n) {
  long long i = (long long)blockIdx.x * blockDim.x + threadIdx.x;
  long long st = (long long)gridDim.x * blockDim.x;
  for (; i < n; i += st) p[i] = v;
}

__global__ void cvt_bf16_kernel(const float* __restrict__ s, __bf16* __restrict__ d, int n) {
  int i = blockIdx.x * blockDim.x + threadIdx.x;
  int st = gridDim.x * blockDim.x;
  for (; i < n; i += st) d[i] = (__bf16)s[i];
}

__global__ void cvt_gelu_bf16_kernel(const float* __restrict__ s, __bf16* __restrict__ d, int n) {
  int i = blockIdx.x * blockDim.x + threadIdx.x;
  int st = gridDim.x * blockDim.x;
  for (; i < n; i += st) d[i] = (__bf16)gelu_exact(s[i]);
}

// transpose-convert: s is [K,N] fp32 row-major -> d is [N,K] bf16 row-major
__global__ void cvt_t_bf16_kernel(const float* __restrict__ s, __bf16* __restrict__ d, int K, int N) {
  int i = blockIdx.x * blockDim.x + threadIdx.x;
  int st = gridDim.x * blockDim.x;
  for (; i < K * N; i += st) {
    int k = i / N, n = i - k * N;
    d[(size_t)n * K + k] = (__bf16)s[i];
  }
}

// C[M,N] = epilogue( Abf[M,K] @ WtBf[N,K]^T + bias[N] )
// One 16-wide N strip per BLOCK: weight strip (16 x K bf16, <=4KB) is DMA'd into LDS
// by the Tensor Data Mover (wave 0 issues tensor_load_to_lds, waits TENSORcnt, block barrier),
// then all 8 waves read B fragments from LDS while streaming A fragments from global.
__global__ __launch_bounds__(256)
void gemm_bf16_kernel(const __bf16* __restrict__ A, const __bf16* __restrict__ Wt,
                      const float* __restrict__ bias,
                      float* __restrict__ Cout, __bf16* __restrict__ Cbf,
                      const float* __restrict__ Prev, const float* __restrict__ gate,
                      int M, int N, int K, int flags)
{
  __shared__ __bf16 wlds[16 * 128];               // one weight strip, K <= 128
  const int lane = threadIdx.x & 31;
  const int wave = threadIdx.x >> 5;
  const int nt = N >> 4;
  const int blockM = blockIdx.x / nt;
  const int tn = blockIdx.x - blockM * nt;
  const int n0 = tn << 4;
  const int mt16 = M >> 4;

#if __has_builtin(__builtin_amdgcn_tensor_load_to_lds)
  if (wave == 0) {
    // Tensor DMA descriptor (ISA ch.8): 2D tile, data_size=2B, tile = 16 rows x K elems.
    unsigned ldsOff = (unsigned)(size_t)(&wlds[0]);
    unsigned long long ga = (unsigned long long)(size_t)(Wt + (size_t)n0 * K);
    v4u g0;
    g0.x = 1u;                                              // count=1 (valid), user mode
    g0.y = ldsOff;                                          // lds_addr
    g0.z = (unsigned)(ga & 0xffffffffu);                    // global_addr[31:0]
    g0.w = ((unsigned)(ga >> 32) & 0x01ffffffu) | (2u << 30); // global_addr[56:32] | type=2
    v8i g1 = { (int)(1u << 16),                             // data_size = 1 (2 bytes)
               (int)(((unsigned)K & 0xffffu) << 16),        // tensor_dim0 = K (bits 79:48)
               (int)(16u << 16),                            // tensor_dim1 = 16 (bits 111:80)
               (int)(((unsigned)K & 0xffffu) << 16),        // tile_dim0 = K (bits 127:112)
               16,                                          // tile_dim1 = 16
               K,                                           // tensor_dim0_stride = K
               0, 0 };
    v4i g2 = {0, 0, 0, 0};
    v4i g3 = {0, 0, 0, 0};
#if __clang_major__ >= 23
    v8i g4 = {0, 0, 0, 0, 0, 0, 0, 0};
    __builtin_amdgcn_tensor_load_to_lds(g0, g1, g2, g3, g4, 0);
#else
    __builtin_amdgcn_tensor_load_to_lds(g0, g1, g2, g3, 0);
#endif
    __builtin_amdgcn_s_wait_tensorcnt(0);
  }
#else
  for (int idx = threadIdx.x; idx < 16 * K; idx += 256)
    wlds[idx] = Wt[(size_t)n0 * K + idx];
#endif
  __syncthreads();

  const int half = lane >> 4, r = lane & 15;
  const int tmBase = blockM * 32 + wave * 4;       // 4 consecutive M-tiles per wave
  if (tmBase >= mt16) return;                      // after barrier: safe

  v8f acc[4];
#pragma unroll
  for (int i = 0; i < 4; ++i) acc[i] = (v8f){0.f,0.f,0.f,0.f,0.f,0.f,0.f,0.f};

  const __bf16* wrow = wlds + r * K;               // LDS row of weight column n0+r
  for (int k0 = 0; k0 < K; k0 += 32) {
    v8bf blo = *(const v8bf*)(wrow + k0 + half * 8);
    v8bf bhi = *(const v8bf*)(wrow + k0 + 16 + half * 8);
    v16bf b = CAT16(blo, bhi);
    v16bf afr[4];
#pragma unroll
    for (int i = 0; i < 4; ++i) {                  // preload all A fragments first
      int mrow = (tmBase + i) * 16 + r;
      if (mrow >= M) mrow = M - 1;
      const __bf16* arow = A + (size_t)mrow * K;
      __builtin_prefetch(arow + k0 + 64, 0, 0);
      v8bf alo = *(const v8bf*)(arow + k0 + half * 8);
      v8bf ahi = *(const v8bf*)(arow + k0 + 16 + half * 8);
      afr[i] = CAT16(alo, ahi);
    }
#pragma unroll
    for (int i = 0; i < 4; ++i)
      acc[i] = __builtin_amdgcn_wmma_f32_16x16x32_bf16(false, afr[i], false, b, (short)0, acc[i], false, false);
  }

  const int n = n0 + r;
  const float bv = bias ? bias[n] : 0.f;
  float g = 0.f;
  if (flags & GF_SKIP) g = 1.f / (1.f + expf(-gate[0]));
#pragma unroll
  for (int i = 0; i < 4; ++i) {
    if (tmBase + i >= mt16) continue;
#pragma unroll
    for (int v = 0; v < 8; ++v) {
      int m = (tmBase + i) * 16 + v + 8 * half;
      float val = acc[i][v] + bv;
      if (flags & GF_RELU) val = fmaxf(val, 0.f);
      if (flags & GF_SKIP) val = g * val + (1.f - g) * Prev[(size_t)m * N + n];
      if (Cout) Cout[(size_t)m * N + n] = val;
      if (Cbf)  Cbf[(size_t)m * N + n] = (__bf16)val;
    }
  }
}

// Pass 1: kt = k[src] @ arel[h] via WMMA (K=16 zero-padded to 32, arel staged transposed in LDS),
// al[e,h] = prel[h]/sqrt(D) * <q[dst,h,:], kt[e,:]>; atomicMax per-(dst,h) running max.
__global__ __launch_bounds__(256)
void edge_logits_kernel(const __bf16* __restrict__ kqvS, const __bf16* __restrict__ kqvD,
                        const int* __restrict__ ei, const float* __restrict__ arel,
                        const float* __restrict__ prel, float* __restrict__ al,
                        unsigned* __restrict__ mord, int E)
{
  __shared__ __bf16 relT[HHE * DD * DD];     // [h][f][d] (transposed), 4 KB
  for (int idx = threadIdx.x; idx < HHE * DD * DD; idx += 256) {
    int h = idx >> 8, rem = idx & 255, d2 = rem >> 4, f = rem & 15;
    relT[h * 256 + f * 16 + d2] = (__bf16)arel[h * 256 + d2 * 16 + f];
  }
  __syncthreads();

  const int lane = threadIdx.x & 31;
  const int wave = threadIdx.x >> 5;
  const int nTiles = (E + 15) >> 4;
  int tile = blockIdx.x * 8 + wave;
  if (tile >= nTiles) return;
  const int half = lane >> 4, r = lane & 15;
  int eIdx = tile * 16 + r; if (eIdx >= E) eIdx = E - 1;
  const int src  = ei[eIdx];
  const int dstR = ei[E + eIdx];
  const v8bf zb = {(__bf16)0.f,(__bf16)0.f,(__bf16)0.f,(__bf16)0.f,
                   (__bf16)0.f,(__bf16)0.f,(__bf16)0.f,(__bf16)0.f};

#pragma unroll
  for (int h = 0; h < HHE; ++h) {
    v8bf alo = *(const v8bf*)(kqvS + (size_t)src * (3 * CDIM) + h * DD + half * 8);
    v8bf blo = *(const v8bf*)(&relT[h * 256 + r * 16 + half * 8]);
    v16bf a = CAT16(alo, zb);
    v16bf b = CAT16(blo, zb);
    v8f acc = {0.f,0.f,0.f,0.f,0.f,0.f,0.f,0.f};
    acc = __builtin_amdgcn_wmma_f32_16x16x32_bf16(false, a, false, b, (short)0, acc, false, false);

    const float scale = prel[h] * 0.25f;     // 1/sqrt(16)
    float myAl = 0.f;
#pragma unroll
    for (int v = 0; v < 8; ++v) {
      int m = v + 8 * half;                  // row of kt this lane holds
      int dm = __shfl(dstR, m, 32);
      float qv = (float)kqvD[(size_t)dm * (3 * CDIM) + CDIM + h * DD + r];
      float t = qv * acc[v];
      t += __shfl_xor(t, 1, 32);
      t += __shfl_xor(t, 2, 32);
      t += __shfl_xor(t, 4, 32);
      t += __shfl_xor(t, 8, 32);             // reduce over 16 cols, stays in half
      if (r == v) myAl = t * scale;
    }
    int mOwn = r + 8 * half;
    int dOwn = __shfl(dstR, mOwn & 15, 32);
    int eOwn = tile * 16 + mOwn;
    if (r < 8 && eOwn < E) {
      al[(size_t)eOwn * HHE + h] = myAl;
      atomicMax(&mord[(size_t)dOwn * HHE + h], ford(myAl));
    }
  }
}

// Pass 2: al <- exp(al - max[dst,h]); atomicAdd per-(dst,h) sums.
__global__ void edge_exp_kernel(float* __restrict__ al, const int* __restrict__ ei,
                                const unsigned* __restrict__ mord, float* __restrict__ ssum, int E)
{
  int idx = blockIdx.x * blockDim.x + threadIdx.x;
  if (idx >= E * HHE) return;
  int e = idx >> 3, h = idx & 7;
  int dst = ei[E + e];
  float m = funord(mord[(size_t)dst * HHE + h]);
  if (!(m > -1e30f && m < 1e30f)) m = 0.f;
  float w = expf(al[idx] - m);
  al[idx] = w;
  atomicAdd(&ssum[(size_t)dst * HHE + h], w);
}

// Pass 3: vt = v[src] @ mrel[h] via WMMA (mrel staged transposed in LDS);
// agg[dst,h,:] += vt * al/(sum+eps).
__global__ __launch_bounds__(256)
void edge_agg_kernel(const __bf16* __restrict__ kqvS, const int* __restrict__ ei,
                     const float* __restrict__ mrel, const float* __restrict__ al,
                     const float* __restrict__ ssum, float* __restrict__ agg, int E)
{
  __shared__ __bf16 relT[HHE * DD * DD];
  for (int idx = threadIdx.x; idx < HHE * DD * DD; idx += 256) {
    int h = idx >> 8, rem = idx & 255, d2 = rem >> 4, f = rem & 15;
    relT[h * 256 + f * 16 + d2] = (__bf16)mrel[h * 256 + d2 * 16 + f];
  }
  __syncthreads();

  const int lane = threadIdx.x & 31;
  const int wave = threadIdx.x >> 5;
  const int nTiles = (E + 15) >> 4;
  int tile = blockIdx.x * 8 + wave;
  if (tile >= nTiles) return;
  const int half = lane >> 4, r = lane & 15;
  int eIdx = tile * 16 + r; if (eIdx >= E) eIdx = E - 1;
  const int src  = ei[eIdx];
  const int dstR = ei[E + eIdx];
  const v8bf zb = {(__bf16)0.f,(__bf16)0.f,(__bf16)0.f,(__bf16)0.f,
                   (__bf16)0.f,(__bf16)0.f,(__bf16)0.f,(__bf16)0.f};

#pragma unroll
  for (int h = 0; h < HHE; ++h) {
    v8bf alo = *(const v8bf*)(kqvS + (size_t)src * (3 * CDIM) + 2 * CDIM + h * DD + half * 8);
    v8bf blo = *(const v8bf*)(&relT[h * 256 + r * 16 + half * 8]);
    v16bf a = CAT16(alo, zb);
    v16bf b = CAT16(blo, zb);
    v8f acc = {0.f,0.f,0.f,0.f,0.f,0.f,0.f,0.f};
    acc = __builtin_amdgcn_wmma_f32_16x16x32_bf16(false, a, false, b, (short)0, acc, false, false);
#pragma unroll
    for (int v = 0; v < 8; ++v) {
      int m  = v + 8 * half;
      int dm = __shfl(dstR, m, 32);
      int e2 = tile * 16 + m;
      if (e2 < E) {
        float w = al[(size_t)e2 * HHE + h] / (ssum[(size_t)dm * HHE + h] + 1e-16f);
        atomicAdd(&agg[(size_t)dm * CDIM + h * DD + r], acc[v] * w);
      }
    }
  }
}

__global__ void pool_kernel(const float* __restrict__ h, float* __restrict__ pooled,
                            int Nn, int off, float inv)
{
  int c = threadIdx.x;                               // 0..127
  int rows = (Nn + gridDim.x - 1) / gridDim.x;
  int r0 = blockIdx.x * rows;
  int r1 = r0 + rows; if (r1 > Nn) r1 = Nn;
  float acc = 0.f;
  for (int r = r0; r < r1; ++r) acc += h[(size_t)r * CDIM + c];
  atomicAdd(&pooled[off + c], acc * inv);
}

__global__ __launch_bounds__(256)
void head_kernel(const float* __restrict__ pooled,
                 const float* sh1_W, const float* sh1_b,
                 const float* sh2_W, const float* sh2_b,
                 const float* ph0_W, const float* ph0_b,
                 const float* ph1_W, const float* ph1_b,
                 const float* ph2_W, const float* ph2_b,
                 const float* vh1_W, const float* vh1_b,
                 const float* vh2_W, const float* vh2_b,
                 float* __restrict__ out)
{
  __shared__ float sp[256];
  __shared__ float s1[128];
  __shared__ float s2[128];
  int t = threadIdx.x;
  sp[t] = pooled[t];
  __syncthreads();
  if (t < 128) {
    float acc = sh1_b[t];
    for (int i = 0; i < 256; ++i) acc += sp[i] * sh1_W[i * 128 + t];
    s1[t] = fmaxf(acc, 0.f);
  }
  __syncthreads();
  if (t < 128) {
    float acc = sh2_b[t];
    for (int i = 0; i < 128; ++i) acc += s1[i] * sh2_W[i * 128 + t];
    s2[t] = fmaxf(acc, 0.f);
  }
  __syncthreads();
  if (t < 10) {
    float acc = ph0_b[t];
    for (int i = 0; i < 128; ++i) acc += s2[i] * ph0_W[i * 10 + t];
    out[t] = acc;
  } else if (t < 20) {
    int j = t - 10;
    float acc = ph1_b[j];
    for (int i = 0; i < 128; ++i) acc += s2[i] * ph1_W[i * 10 + j];
    out[10 + j] = acc;
  } else if (t < 24) {
    int j = t - 20;
    float acc = ph2_b[j];
    for (int i = 0; i < 128; ++i) acc += s2[i] * ph2_W[i * 4 + j];
    out[20 + j] = acc;
  } else if (t == 24) {
    float acc = vh2_b[0];
    for (int j = 0; j < 64; ++j) {
      float a2 = vh1_b[j];
      for (int i = 0; i < 128; ++i) a2 += s2[i] * vh1_W[i * 64 + j];
      acc += fmaxf(a2, 0.f) * vh2_W[j];
    }
    out[24] = acc;
  }
}

extern "C" void kernel_launch(void* const* d_in, const int* in_sizes, int n_in,
                              void* d_out, int out_size, void* d_ws, size_t ws_size,
                              hipStream_t stream)
{
  const float* xA      = (const float*)d_in[0];
  const float* xB      = (const float*)d_in[1];
  const int*   ei_ab   = (const int*)d_in[2];
  const int*   ei_ba   = (const int*)d_in[3];
  const float* linA_W  = (const float*)d_in[4];
  const float* linA_b  = (const float*)d_in[5];
  const float* linB_W  = (const float*)d_in[6];
  const float* linB_b  = (const float*)d_in[7];
  const float* kqvA_W  = (const float*)d_in[8];
  const float* kqvA_b  = (const float*)d_in[9];
  const float* kqvB_W  = (const float*)d_in[10];
  const float* kqvB_b  = (const float*)d_in[11];
  const float* arel_ab = (const float*)d_in[12];
  const float* mrel_ab = (const float*)d_in[13];
  const float* prel_ab = (const float*)d_in[14];
  const float* arel_ba = (const float*)d_in[15];
  const float* mrel_ba = (const float*)d_in[16];
  const float* prel_ba = (const float*)d_in[17];
  const float* outA_W  = (const float*)d_in[18];
  const float* outA_b  = (const float*)d_in[19];
  const float* outB_W  = (const float*)d_in[20];
  const float* outB_b  = (const float*)d_in[21];
  const float* skipA   = (const float*)d_in[22];
  const float* skipB   = (const float*)d_in[23];
  const float* sh1_W   = (const float*)d_in[24];
  const float* sh1_b   = (const float*)d_in[25];
  const float* sh2_W   = (const float*)d_in[26];
  const float* sh2_b   = (const float*)d_in[27];
  const float* ph0_W   = (const float*)d_in[28];
  const float* ph0_b   = (const float*)d_in[29];
  const float* ph1_W   = (const float*)d_in[30];
  const float* ph1_b   = (const float*)d_in[31];
  const float* ph2_W   = (const float*)d_in[32];
  const float* ph2_b   = (const float*)d_in[33];
  const float* vh1_W   = (const float*)d_in[34];
  const float* vh1_b   = (const float*)d_in[35];
  const float* vh2_W   = (const float*)d_in[36];
  const float* vh2_b   = (const float*)d_in[37];
  float* out = (float*)d_out;

  // workspace carve-up (float units, 1KB aligned chunks)
  float* ws = (float*)d_ws;
  size_t off = 0;
  auto alloc = [&](size_t nFloats) -> float* {
    float* p = ws + off;
    off += (nFloats + 255) & ~(size_t)255;
    return p;
  };
  auto allocBf = [&](size_t nBf) -> __bf16* {
    return (__bf16*)alloc((nBf + 1) / 2);
  };
  float*    hA      = alloc((size_t)NA_N * CDIM);
  float*    hB      = alloc((size_t)NB_N * CDIM);
  float*    aggA    = alloc((size_t)NA_N * CDIM);
  float*    aggB    = alloc((size_t)NB_N * CDIM);
  float*    al      = alloc((size_t)EDG * HHE);
  unsigned* mord    = (unsigned*)alloc((size_t)NA_N * HHE);
  float*    ssum    = alloc((size_t)NA_N * HHE);
  float*    pooled  = alloc(256);
  __bf16*   xA_bf   = allocBf((size_t)NA_N * 64);
  __bf16*   xB_bf   = allocBf((size_t)NB_N * 32);
  __bf16*   hA_bf   = allocBf((size_t)NA_N * CDIM);
  __bf16*   hB_bf   = allocBf((size_t)NB_N * CDIM);
  __bf16*   kqvA_bf = allocBf((size_t)NA_N * 3 * CDIM);
  __bf16*   kqvB_bf = allocBf((size_t)NB_N * 3 * CDIM);
  __bf16*   gbf     = allocBf((size_t)NA_N * CDIM);     // gelu(agg) staging (reused A then B)
  __bf16*   linA_t  = allocBf((size_t)64 * CDIM);
  __bf16*   linB_t  = allocBf((size_t)32 * CDIM);
  __bf16*   kqvA_t  = allocBf((size_t)LL * CDIM * 3 * CDIM);
  __bf16*   kqvB_t  = allocBf((size_t)LL * CDIM * 3 * CDIM);
  __bf16*   outA_t  = allocBf((size_t)LL * CDIM * CDIM);
  __bf16*   outB_t  = allocBf((size_t)LL * CDIM * CDIM);

  auto cvt = [&](const float* s, __bf16* d, int n) {
    cvt_bf16_kernel<<<(n + 255) / 256, 256, 0, stream>>>(s, d, n);
  };
  auto cvtT = [&](const float* s, __bf16* d, int K, int N) {
    cvt_t_bf16_kernel<<<(K * N + 255) / 256, 256, 0, stream>>>(s, d, K, N);
  };

  // weights: transpose + bf16
  cvtT(linA_W, linA_t, 64, CDIM);
  cvtT(linB_W, linB_t, 32, CDIM);
  for (int l = 0; l < LL; ++l) {
    cvtT(kqvA_W + (size_t)l * CDIM * 3 * CDIM, kqvA_t + (size_t)l * CDIM * 3 * CDIM, CDIM, 3 * CDIM);
    cvtT(kqvB_W + (size_t)l * CDIM * 3 * CDIM, kqvB_t + (size_t)l * CDIM * 3 * CDIM, CDIM, 3 * CDIM);
    cvtT(outA_W + (size_t)l * CDIM * CDIM, outA_t + (size_t)l * CDIM * CDIM, CDIM, CDIM);
    cvtT(outB_W + (size_t)l * CDIM * CDIM, outB_t + (size_t)l * CDIM * CDIM, CDIM, CDIM);
  }
  // input activations -> bf16
  cvt(xA, xA_bf, NA_N * 64);
  cvt(xB, xB_bf, NB_N * 32);

  auto gemm = [&](const __bf16* Ap, const __bf16* Wp, const float* bp,
                  float* Cp, __bf16* Cbfp, const float* Pp, const float* gp,
                  int M, int N, int K, int flags) {
    int mt16 = M / 16;
    int gridM = (mt16 + 31) / 32;                 // 32 M-tiles (512 rows) per block
    int blocks = gridM * (N / 16);
    gemm_bf16_kernel<<<blocks, 256, 0, stream>>>(Ap, Wp, bp, Cp, Cbfp, Pp, gp, M, N, K, flags);
  };

  gemm(xA_bf, linA_t, linA_b, hA, hA_bf, nullptr, nullptr, NA_N, CDIM, 64, GF_RELU);
  gemm(xB_bf, linB_t, linB_b, hB, hB_bf, nullptr, nullptr, NB_N, CDIM, 32, GF_RELU);

  const int edgeBlocks = ((EDG / 16) + 7) / 8;
  const int expBlocks  = (EDG * HHE + 255) / 256;

  for (int l = 0; l < LL; ++l) {
    gemm(hA_bf, kqvA_t + (size_t)l * CDIM * 3 * CDIM, kqvA_b + l * 3 * CDIM,
         nullptr, kqvA_bf, nullptr, nullptr, NA_N, 3 * CDIM, CDIM, 0);
    gemm(hB_bf, kqvB_t + (size_t)l * CDIM * 3 * CDIM, kqvB_b + l * 3 * CDIM,
         nullptr, kqvB_bf, nullptr, nullptr, NB_N, 3 * CDIM, CDIM, 0);

    // direction A -> B (src side = A, dst side = B)
    fill_u32_kernel<<<640, 256, 0, stream>>>(mord, 0x007fffffu, (long long)NB_N * HHE);
    fill_u32_kernel<<<640, 256, 0, stream>>>((unsigned*)ssum, 0u, (long long)NB_N * HHE);
    fill_u32_kernel<<<2048, 256, 0, stream>>>((unsigned*)aggB, 0u, (long long)NB_N * CDIM);
    edge_logits_kernel<<<edgeBlocks, 256, 0, stream>>>(
        kqvA_bf, kqvB_bf, ei_ab, arel_ab + l * HHE * DD * DD, prel_ab + l * HHE, al, mord, EDG);
    edge_exp_kernel<<<expBlocks, 256, 0, stream>>>(al, ei_ab, mord, ssum, EDG);
    edge_agg_kernel<<<edgeBlocks, 256, 0, stream>>>(
        kqvA_bf, ei_ab, mrel_ab + l * HHE * DD * DD, al, ssum, aggB, EDG);

    // direction B -> A (src side = B, dst side = A)
    fill_u32_kernel<<<640, 256, 0, stream>>>(mord, 0x007fffffu, (long long)NA_N * HHE);
    fill_u32_kernel<<<640, 256, 0, stream>>>((unsigned*)ssum, 0u, (long long)NA_N * HHE);
    fill_u32_kernel<<<2048, 256, 0, stream>>>((unsigned*)aggA, 0u, (long long)NA_N * CDIM);
    edge_logits_kernel<<<edgeBlocks, 256, 0, stream>>>(
        kqvB_bf, kqvA_bf, ei_ba, arel_ba + l * HHE * DD * DD, prel_ba + l * HHE, al, mord, EDG);
    edge_exp_kernel<<<expBlocks, 256, 0, stream>>>(al, ei_ba, mord, ssum, EDG);
    edge_agg_kernel<<<edgeBlocks, 256, 0, stream>>>(
        kqvB_bf, ei_ba, mrel_ba + l * HHE * DD * DD, al, ssum, aggA, EDG);

    // output projection: gelu(agg) (converted to bf16 out-of-line) + skip blend, in place into h
    cvt_gelu_bf16_kernel<<<(NA_N * CDIM + 255) / 256, 256, 0, stream>>>(aggA, gbf, NA_N * CDIM);
    gemm(gbf, outA_t + (size_t)l * CDIM * CDIM, outA_b + l * CDIM, hA, hA_bf,
         hA, skipA + l, NA_N, CDIM, CDIM, GF_SKIP);
    cvt_gelu_bf16_kernel<<<(NB_N * CDIM + 255) / 256, 256, 0, stream>>>(aggB, gbf, NB_N * CDIM);
    gemm(gbf, outB_t + (size_t)l * CDIM * CDIM, outB_b + l * CDIM, hB, hB_bf,
         hB, skipB + l, NB_N, CDIM, CDIM, GF_SKIP);
  }

  fill_u32_kernel<<<1, 256, 0, stream>>>((unsigned*)pooled, 0u, 256);
  pool_kernel<<<160, 128, 0, stream>>>(hA, pooled, NA_N, 0, 1.0f / NA_N);
  pool_kernel<<<160, 128, 0, stream>>>(hB, pooled, NB_N, 128, 1.0f / NB_N);
  head_kernel<<<1, 256, 0, stream>>>(pooled, sh1_W, sh1_b, sh2_W, sh2_b,
                                     ph0_W, ph0_b, ph1_W, ph1_b, ph2_W, ph2_b,
                                     vh1_W, vh1_b, vh2_W, vh2_b, out);
}
